// Attention_54245436948549
// MI455X (gfx1250) — compile-verified, matrix operand order checked by reading, not run on hardware
//
#include <hip/hip_runtime.h>
#include <hip/hip_bf16.h>

typedef __bf16 bf16;
typedef __attribute__((ext_vector_type(16))) __bf16 v16bf;
typedef __attribute__((ext_vector_type(8)))  __bf16 v8bf;
typedef __attribute__((ext_vector_type(8)))  float  v8f;
typedef __attribute__((ext_vector_type(4)))  unsigned int u32x4;
typedef __attribute__((ext_vector_type(4)))  int          i32x4;
typedef __attribute__((ext_vector_type(8)))  int          i32x8;

typedef __attribute__((address_space(1))) i32x4 gi32x4;   // global int4
typedef __attribute__((address_space(3))) i32x4 li32x4;   // LDS int4
typedef __attribute__((address_space(3))) __bf16 lbf16;

#define BATCH 4
#define SEQ   2048
#define DMODEL 1024
#define NHEAD 16
#define HDIM  64
#define MTOT  (BATCH * SEQ)        // 8192
#define N_QKV (3 * DMODEL)         // 3072

#ifndef __has_builtin
#define __has_builtin(x) 0
#endif

#if __has_builtin(__builtin_amdgcn_global_load_async_to_lds_b128)
#define HAVE_ASYNC 1
#else
#define HAVE_ASYNC 0
#endif

#if __has_builtin(__builtin_amdgcn_tensor_load_to_lds) && \
    __has_builtin(__builtin_amdgcn_s_wait_tensorcnt)
#define HAVE_TDM 1
#else
#define HAVE_TDM 0
#endif

// ---------------------------------------------------------------------------
// Async global->LDS 16-byte copy (ASYNCcnt path), with sync fallback.
// Builtin signature (from hipcc diagnostic): (AS1 int4*, AS3 int4*, imm, imm)
// ---------------------------------------------------------------------------
__device__ __forceinline__ void async_copy16(const bf16* g, bf16* l) {
#if HAVE_ASYNC
  __builtin_amdgcn_global_load_async_to_lds_b128((gi32x4*)g, (li32x4*)l, 0, 0);
#else
  *(v8bf*)l = *(const v8bf*)g;
#endif
}

__device__ __forceinline__ void async_wait_all() {
#if HAVE_ASYNC
#if __has_builtin(__builtin_amdgcn_s_wait_asynccnt)
  __builtin_amdgcn_s_wait_asynccnt(0);
#else
  __asm volatile("s_wait_asynccnt 0" ::: "memory");
#endif
#endif
}

// ---------------------------------------------------------------------------
// TDM: 2D tile load global->LDS via tensor descriptor (TENSORcnt path).
// data_size = 2 bytes (bf16).  tile_d0 elements contiguous, tile_d1 lines
// of stride0 elements.  Descriptor bit layout per CDNA5 ISA §8.3/8.4.
// ---------------------------------------------------------------------------
__device__ __forceinline__ void tdm_load_2d_bf16(unsigned lds_off, const void* gptr,
                                                 unsigned tile_d0, unsigned tile_d1,
                                                 unsigned tensor_d0, unsigned tensor_d1,
                                                 unsigned stride0) {
#if HAVE_TDM
  unsigned long long ga = (unsigned long long)(size_t)gptr;
  u32x4 g0;
  g0[0] = 1u;                                         // count=1, user mode
  g0[1] = lds_off;                                    // LDS byte address
  g0[2] = (unsigned)ga;                               // global_addr[31:0]
  g0[3] = (unsigned)(ga >> 32) | (2u << 30);          // global_addr[56:32] | type=2
  i32x8 g1;
  g1[0] = (int)(1u << 16);                            // data_size=1 (2 bytes)
  g1[1] = (int)(tensor_d0 << 16);                     // tensor_dim0[15:0]
  g1[2] = (int)((tensor_d0 >> 16) | (tensor_d1 << 16));
  g1[3] = (int)((tensor_d1 >> 16) | (tile_d0 << 16)); // tile_dim0
  g1[4] = (int)(tile_d1 & 0xffffu);                   // tile_dim1 (tile_dim2=0)
  g1[5] = (int)stride0;                               // tensor_dim0_stride[31:0]
  g1[6] = 0;
  g1[7] = 0;
  i32x4 z4 = {0, 0, 0, 0};
#if __clang_major__ >= 23
  i32x8 z8 = {0, 0, 0, 0, 0, 0, 0, 0};
  __builtin_amdgcn_tensor_load_to_lds(g0, g1, z4, z4, z8, 0);
#else
  __builtin_amdgcn_tensor_load_to_lds(g0, g1, z4, z4, 0);
#endif
#endif
}

// ---------------------------------------------------------------------------
// WMMA helper: D = A(16x32 bf16) * B(32x16 bf16) + C(16x16 f32)
// ---------------------------------------------------------------------------
__device__ __forceinline__ v8f wmma_bf16(v16bf a, v16bf b, v8f c) {
  return __builtin_amdgcn_wmma_f32_16x16x32_bf16(
      false, a, false, b, (short)0, c, false, false);
}

// 16x32 operand fragment (row-major, ld multiple of 8, contraction contiguous).
__device__ __forceinline__ v16bf load_frag(const bf16* p, int ld, int row0, int k0) {
  const int l = threadIdx.x & 31;
  const bf16* q = p + (size_t)(row0 + (l & 15)) * ld + (k0 + ((l >> 4) << 3));
  v8bf lo = *(const v8bf*)q;
  v8bf hi = *(const v8bf*)(q + 16);
  return __builtin_shufflevector(lo, hi, 0,1,2,3,4,5,6,7,8,9,10,11,12,13,14,15);
}

// ---------------------------------------------------------------------------
// Kernel 0: convert x -> bf16; build transposed bf16 weights Wt[n][k]
// ---------------------------------------------------------------------------
__global__ void __launch_bounds__(256)
convert_kernel(const float* __restrict__ x, const float* __restrict__ Wq,
               const float* __restrict__ Wp, bf16* __restrict__ xbf,
               bf16* __restrict__ Wqt, bf16* __restrict__ Wpt) {
  size_t i = (size_t)blockIdx.x * 256 + threadIdx.x;
  xbf[i] = (bf16)x[i];
  if (i < (size_t)N_QKV * DMODEL) {
    int n = (int)(i >> 10), k = (int)(i & 1023);
    Wqt[i] = (bf16)Wq[(size_t)k * N_QKV + n];
  }
  if (i < (size_t)DMODEL * DMODEL) {
    int n = (int)(i >> 10), k = (int)(i & 1023);
    Wpt[i] = (bf16)Wp[(size_t)k * DMODEL + n];
  }
}

// ---------------------------------------------------------------------------
// GEMM mainloop with double-buffered ASYNC global->LDS staging.
// As/Bs are 2 x (128x32) bf16 buffers.  8 waves, per-wave 64x32 via 4x2 WMMAs.
// ---------------------------------------------------------------------------
template <int KDIM>
__device__ __forceinline__ void gemm_mainloop(const bf16* __restrict__ A,
                                              const bf16* __restrict__ Bt,
                                              bf16* As, bf16* Bs,
                                              int m0, int n0, v8f acc[4][2]) {
  const int t  = threadIdx.x;
  const int w  = t >> 5;
  const int wm = (w & 1) * 64;
  const int wn = (w >> 1) * 32;
  const int row = t >> 1;
  const int cb  = (t & 1) * 16;
  const bf16* ag = A  + (size_t)(m0 + row) * KDIM + cb;
  const bf16* bg = Bt + (size_t)(n0 + row) * KDIM + cb;
  const int lofs = row * 32 + cb;

  // prologue: stage k-step 0 into buffer 0
  async_copy16(ag,     As + lofs);
  async_copy16(ag + 8, As + lofs + 8);
  async_copy16(bg,     Bs + lofs);
  async_copy16(bg + 8, Bs + lofs + 8);

  int buf = 0;
  for (int k0 = 0; k0 < KDIM; k0 += 32) {
    async_wait_all();      // our tile copies have landed in LDS
    __syncthreads();       // everyone's copies landed; prev buffer reads done

    bf16* Ac = As + buf * (128 * 32);
    bf16* Bc = Bs + buf * (128 * 32);
    if (k0 + 32 < KDIM) {  // prefetch next k-step into the other buffer
      bf16* An = As + (buf ^ 1) * (128 * 32);
      bf16* Bn = Bs + (buf ^ 1) * (128 * 32);
      async_copy16(ag + k0 + 32,     An + lofs);
      async_copy16(ag + k0 + 40,     An + lofs + 8);
      async_copy16(bg + k0 + 32,     Bn + lofs);
      async_copy16(bg + k0 + 40,     Bn + lofs + 8);
    }

    v16bf af[4], bfv[2];
#pragma unroll
    for (int i = 0; i < 4; i++) af[i]  = load_frag(Ac, 32, wm + i * 16, 0);
#pragma unroll
    for (int j = 0; j < 2; j++) bfv[j] = load_frag(Bc, 32, wn + j * 16, 0);
#pragma unroll
    for (int i = 0; i < 4; i++)
#pragma unroll
      for (int j = 0; j < 2; j++)
        acc[i][j] = wmma_bf16(af[i], bfv[j], acc[i][j]);
    buf ^= 1;
  }
}

// ---------------------------------------------------------------------------
// Kernel 1: QKV projection -> Q/K/V[B,H,N,64] bf16, Q pre-scaled by 0.125
// ---------------------------------------------------------------------------
__global__ void __launch_bounds__(256)
qkv_gemm_kernel(const bf16* __restrict__ xbf, const bf16* __restrict__ Wqt,
                const float* __restrict__ bqkv, bf16* __restrict__ Qb,
                bf16* __restrict__ Kb, bf16* __restrict__ Vb) {
  __shared__ __align__(16) bf16 As[2 * 128 * 32];
  __shared__ __align__(16) bf16 Bs[2 * 128 * 32];
  const int m0 = blockIdx.y * 128;
  const int n0 = blockIdx.x * 128;
  v8f acc[4][2] = {};
  gemm_mainloop<DMODEL>(xbf, Wqt, As, Bs, m0, n0, acc);

  const int l = threadIdx.x & 31, w = threadIdx.x >> 5;
  const int wm = (w & 1) * 64, wn = (w >> 1) * 32;
#pragma unroll
  for (int i = 0; i < 4; i++)
#pragma unroll
    for (int j = 0; j < 2; j++)
#pragma unroll
      for (int e = 0; e < 8; e++) {
        int m = m0 + wm + i * 16 + ((l >> 4) << 3) + e;
        int c = n0 + wn + j * 16 + (l & 15);
        float v = acc[i][j][e] + bqkv[c];
        int s = c >> 10, h = (c >> 6) & 15, hd = c & 63;
        int bb = m >> 11, np = m & 2047;
        bf16* dst = (s == 0) ? Qb : ((s == 1) ? Kb : Vb);
        if (s == 0) v *= 0.125f;  // fold softmax scale into Q
        dst[(((size_t)(bb * NHEAD + h)) * SEQ + np) * HDIM + hd] = (bf16)v;
      }
}

// ---------------------------------------------------------------------------
// Kernel 2: flash attention.  K tile staged by TDM (tensor_load_to_lds),
// V tile transposed by VALU scatter; S=Q*K^T and O+=P*V via WMMA.
// ---------------------------------------------------------------------------
__global__ void __launch_bounds__(256)
attn_kernel(const bf16* __restrict__ Qb, const bf16* __restrict__ Kb,
            const bf16* __restrict__ Vb, bf16* __restrict__ Obf) {
  __shared__ __align__(16) bf16  Ks[32 * 64];    // [key][hd]
  __shared__ __align__(16) bf16  Vts[64 * 32];   // [hd][key] (transposed)
  __shared__ __align__(16) float Ss[8 * 16 * 32];
  __shared__ __align__(16) bf16  Ps[8 * 16 * 32];
  __shared__ float rt[8][16];

  const int l = threadIdx.x & 31, w = threadIdx.x >> 5;
  const int r = l & 15, half = l >> 4;
  const int qt = blockIdx.x & 15;
  const int bh = blockIdx.x >> 4;
  const int b  = bh >> 4, h = bh & 15;
  const bf16* Qp = Qb + (size_t)bh * (SEQ * HDIM);
  const bf16* Kp = Kb + (size_t)bh * (SEQ * HDIM);
  const bf16* Vp = Vb + (size_t)bh * (SEQ * HDIM);
  const int qr0 = qt * 128 + w * 16;

#if HAVE_TDM
  const unsigned ks_off = (unsigned)(size_t)(lbf16*)Ks;
#endif

  v16bf aq[2];
#pragma unroll
  for (int c = 0; c < 2; c++) aq[c] = load_frag(Qp, HDIM, qr0, c * 32);

  v8f vo[4] = {};
  float mi = -__builtin_inff(), li = 0.0f;
  float* sw = Ss + w * (16 * 32);
  bf16*  pw = Ps + w * (16 * 32);

  for (int kt = 0; kt < SEQ; kt += 32) {
    __syncthreads();  // prior reads of Ks/Vts complete

#if HAVE_TDM
    if (w == 0)  // one wave drives the TDM for the whole 32x64 bf16 K tile
      tdm_load_2d_bf16(ks_off, Kp + (size_t)kt * HDIM,
                       /*tile_d0=*/HDIM, /*tile_d1=*/32,
                       /*tensor_d0=*/HDIM, /*tensor_d1=*/SEQ - kt,
                       /*stride0=*/HDIM);
#endif
    {  // V transpose staging (and K staging on the fallback path)
      const int t = threadIdx.x;
      const int key = t >> 3, hd0 = (t & 7) * 8;
#if !HAVE_TDM
      v8bf kv = *(const v8bf*)(Kp + (size_t)(kt + key) * HDIM + hd0);
      *(v8bf*)(Ks + key * 64 + hd0) = kv;
#endif
      v8bf vv = *(const v8bf*)(Vp + (size_t)(kt + key) * HDIM + hd0);
#pragma unroll
      for (int e = 0; e < 8; e++) Vts[(hd0 + e) * 32 + key] = vv[e];
    }
#if HAVE_TDM
    if (w == 0) __builtin_amdgcn_s_wait_tensorcnt(0);
#endif
    __syncthreads();

    // S = Q * K^T  (16 q-rows x 32 keys), contraction over hd in two chunks
    v8f s0 = {}, s1 = {};
#pragma unroll
    for (int c = 0; c < 2; c++) {
      v16bf b0 = load_frag(Ks, 64, 0,  c * 32);
      v16bf b1 = load_frag(Ks, 64, 16, c * 32);
      s0 = wmma_bf16(aq[c], b0, s0);
      s1 = wmma_bf16(aq[c], b1, s1);
    }
#pragma unroll
    for (int e = 0; e < 8; e++) {
      sw[(half * 8 + e) * 32 + r]      = s0[e];
      sw[(half * 8 + e) * 32 + 16 + r] = s1[e];
    }
    __asm volatile("s_wait_dscnt 0" ::: "memory");

    // online softmax: lane (r, half) reduces its half-row of 16
    const float* myrow = sw + r * 32 + half * 16;
    float tm = myrow[0];
#pragma unroll
    for (int i = 1; i < 16; i++) tm = fmaxf(tm, myrow[i]);
    tm = fmaxf(tm, __shfl_xor(tm, 16));
    float mnew  = fmaxf(mi, tm);
    float alpha = __expf(mi - mnew);
    float psum  = 0.0f;
    bf16* myp = pw + r * 32 + half * 16;
#pragma unroll
    for (int i = 0; i < 16; i++) {
      float p = __expf(myrow[i] - mnew);
      psum += p;
      myp[i] = (bf16)p;
    }
    psum += __shfl_xor(psum, 16);
    li = li * alpha + psum;
    mi = mnew;
    if (half == 0) rt[w][r] = alpha;
    __asm volatile("s_wait_dscnt 0" ::: "memory");

    float a8[8];
#pragma unroll
    for (int e = 0; e < 8; e++) a8[e] = rt[w][half * 8 + e];
#pragma unroll
    for (int tI = 0; tI < 4; tI++)
#pragma unroll
      for (int e = 0; e < 8; e++) vo[tI][e] *= a8[e];

    v16bf ap = load_frag(pw, 32, 0, 0);
#pragma unroll
    for (int tI = 0; tI < 4; tI++) {
      v16bf bv = load_frag(Vts, 32, tI * 16, 0);
      vo[tI] = wmma_bf16(ap, bv, vo[tI]);
    }
  }

  if (half == 0) rt[w][r] = 1.0f / li;
  __asm volatile("s_wait_dscnt 0" ::: "memory");
  float i8[8];
#pragma unroll
  for (int e = 0; e < 8; e++) i8[e] = rt[w][half * 8 + e];
#pragma unroll
  for (int tI = 0; tI < 4; tI++) {
    int hd = tI * 16 + (l & 15);
#pragma unroll
    for (int e = 0; e < 8; e++) {
      int np = qr0 + half * 8 + e;
      float v = vo[tI][e] * i8[e];
      Obf[((size_t)(b * SEQ + np)) * DMODEL + h * HDIM + hd] = (bf16)v;
    }
  }
}

// ---------------------------------------------------------------------------
// Kernel 3: output projection.  out = O @ W_proj + b_proj  (f32 output)
// ---------------------------------------------------------------------------
__global__ void __launch_bounds__(256)
proj_gemm_kernel(const bf16* __restrict__ Obf, const bf16* __restrict__ Wpt,
                 const float* __restrict__ bproj, float* __restrict__ out) {
  __shared__ __align__(16) bf16 As[2 * 128 * 32];
  __shared__ __align__(16) bf16 Bs[2 * 128 * 32];
  const int m0 = blockIdx.y * 128;
  const int n0 = blockIdx.x * 128;
  v8f acc[4][2] = {};
  gemm_mainloop<DMODEL>(Obf, Wpt, As, Bs, m0, n0, acc);

  const int l = threadIdx.x & 31, w = threadIdx.x >> 5;
  const int wm = (w & 1) * 64, wn = (w >> 1) * 32;
#pragma unroll
  for (int i = 0; i < 4; i++)
#pragma unroll
    for (int j = 0; j < 2; j++)
#pragma unroll
      for (int e = 0; e < 8; e++) {
        int m = m0 + wm + i * 16 + ((l >> 4) << 3) + e;
        int c = n0 + wn + j * 16 + (l & 15);
        out[(size_t)m * DMODEL + c] = acc[i][j][e] + bproj[c];
      }
}

// ---------------------------------------------------------------------------
// Host launcher
// ---------------------------------------------------------------------------
extern "C" void kernel_launch(void* const* d_in, const int* in_sizes, int n_in,
                              void* d_out, int out_size, void* d_ws, size_t ws_size,
                              hipStream_t stream) {
  const float* x  = (const float*)d_in[0];
  const float* Wq = (const float*)d_in[1];
  const float* bq = (const float*)d_in[2];
  const float* Wp = (const float*)d_in[3];
  const float* bp = (const float*)d_in[4];
  float* out = (float*)d_out;

  char* ws = (char*)d_ws;
  size_t off = 0;
  auto carve = [&](size_t bytes) {
    char* p = ws + off;
    off = (off + bytes + 255) & ~(size_t)255;
    return p;
  };
  bf16* xbf = (bf16*)carve((size_t)MTOT * DMODEL * 2);
  bf16* Wqt = (bf16*)carve((size_t)N_QKV * DMODEL * 2);
  bf16* Wpt = (bf16*)carve((size_t)DMODEL * DMODEL * 2);
  bf16* Qb  = (bf16*)carve((size_t)BATCH * NHEAD * SEQ * HDIM * 2);
  bf16* Kb  = (bf16*)carve((size_t)BATCH * NHEAD * SEQ * HDIM * 2);
  bf16* Vb  = (bf16*)carve((size_t)BATCH * NHEAD * SEQ * HDIM * 2);
  bf16* Obf = (bf16*)carve((size_t)MTOT * DMODEL * 2);

  convert_kernel<<<(MTOT * DMODEL) / 256, 256, 0, stream>>>(x, Wq, Wp, xbf, Wqt, Wpt);
  qkv_gemm_kernel<<<dim3(N_QKV / 128, MTOT / 128), 256, 0, stream>>>(xbf, Wqt, bq, Qb, Kb, Vb);
  attn_kernel<<<BATCH * NHEAD * (SEQ / 128), 256, 0, stream>>>(Qb, Kb, Vb, Obf);
  proj_gemm_kernel<<<dim3(DMODEL / 128, MTOT / 128), 256, 0, stream>>>(Obf, Wpt, bp, out);
}
